// MultiHeadAttention_56504589746218
// MI455X (gfx1250) — compile-verified, matrix operand order checked by reading, not run on hardware
//
#include <hip/hip_runtime.h>
#include <hip/hip_bf16.h>

// ---------------------------------------------------------------------------
// MI455X (gfx1250) multi-head attention block:
//   out = (attn(LN(q)Wq, kWk, vWv)) Wo^T + q
// All matmuls on v_wmma_f32_16x16x32_f16 (f16 in, f32 accumulate).
// Attention stages K/V tiles into LDS with global_load_async_to_lds_b128
// (ASYNCcnt), double-buffered across the key loop.
// ---------------------------------------------------------------------------

typedef __attribute__((ext_vector_type(16))) _Float16 v16h;
typedef __attribute__((ext_vector_type(8)))  _Float16 v8h;
typedef __attribute__((ext_vector_type(8)))  float    v8f;

#define D_MODEL 1024
#define LSEQ    2048
#define NHEAD   16
#define DHEAD   64
#define BATCH   2
#define NROWS   (BATCH * LSEQ)      // 4096

__device__ __forceinline__ v8f wmma16x16x32(v16h a, v16h b, v8f c) {
  // (neg_a, A, neg_b, B, c_mod, C, reuse_a, reuse_b)
  return __builtin_amdgcn_wmma_f32_16x16x32_f16(false, a, false, b,
                                                (short)0, c, false, false);
}

// Load a 16x32 f16 fragment in the documented gfx1250 A-matrix layout:
//   lane&15 selects the row, lane>>4 selects which K-interleave half;
//   per lane: halves K = {8h..8h+7} and {16+8h..16+8h+7} (two 16B chunks).
// The B-operand (32x16, K x N) uses the symmetric layout with lane&15 = column,
// which for x @ W^T means each lane reads a contiguous piece of ROW n of W.
__device__ __forceinline__ v16h load_frag(const _Float16* __restrict__ base,
                                          int ld, int row0, int k0, int lane) {
  int r = row0 + (lane & 15);
  int h = lane >> 4;
  const _Float16* p = base + (size_t)r * ld + k0 + 8 * h;
  v8h lo = *(const v8h*)p;
  v8h hi = *(const v8h*)(p + 16);
  return __builtin_shufflevector(lo, hi, 0, 1, 2, 3, 4, 5, 6, 7,
                                 8, 9, 10, 11, 12, 13, 14, 15);
}

// Async DMA of one 16B chunk global -> LDS (bypasses VGPRs, ASYNCcnt-tracked).
// For generic pointers to LDS the low 32 bits hold the LDS-relative offset
// (ISA 10.2).
__device__ __forceinline__ void async_b128(const _Float16* lds_dst,
                                           const _Float16* gsrc) {
  unsigned lds_off = (unsigned)(uintptr_t)lds_dst;
  asm volatile("global_load_async_to_lds_b128 %0, %1, off"
               :: "v"(lds_off), "v"(gsrc) : "memory");
}

// ---------------------------------------------------------------------------
// LayerNorm(q) -> f16   (one 256-thread block per row of 1024)
// ---------------------------------------------------------------------------
__global__ __launch_bounds__(256) void ln_kernel(const float* __restrict__ x,
                                                 const float* __restrict__ g,
                                                 const float* __restrict__ bt,
                                                 _Float16* __restrict__ out) {
  int row = blockIdx.x;
  const float* xr = x + (size_t)row * D_MODEL;
  __shared__ float redS[8], redS2[8];
  int lane = threadIdx.x & 31, wave = threadIdx.x >> 5;

  float s = 0.f, s2 = 0.f;
  for (int i = threadIdx.x; i < D_MODEL; i += 256) {
    float v = xr[i];
    s += v; s2 += v * v;
  }
#pragma unroll
  for (int d = 1; d < 32; d <<= 1) {
    s  += __shfl_xor(s, d, 32);
    s2 += __shfl_xor(s2, d, 32);
  }
  if (lane == 0) { redS[wave] = s; redS2[wave] = s2; }
  __syncthreads();
  s = 0.f; s2 = 0.f;
#pragma unroll
  for (int wv = 0; wv < 8; ++wv) { s += redS[wv]; s2 += redS2[wv]; }
  float mu = s * (1.0f / D_MODEL);
  float var = s2 * (1.0f / D_MODEL) - mu * mu;
  float rstd = rsqrtf(var + 1e-6f);
  for (int i = threadIdx.x; i < D_MODEL; i += 256) {
    float v = (xr[i] - mu) * rstd * g[i] + bt[i];
    out[(size_t)row * D_MODEL + i] = (_Float16)v;
  }
}

// ---------------------------------------------------------------------------
// f32 -> f16 elementwise
// ---------------------------------------------------------------------------
__global__ void f32_to_f16(const float* __restrict__ in,
                           _Float16* __restrict__ out, int n) {
  int i = blockIdx.x * blockDim.x + threadIdx.x;
  if (i < n) out[i] = (_Float16)in[i];
}

// ---------------------------------------------------------------------------
// C[m,n] = scale * sum_k X[m,k] * W[n,k]   (x @ W^T), WMMA f16->f32
// Block = 256 thr = 8 waves (4 along M x 2 along N); wave tile = 32x64.
// modes: 0 = Q store  [B,H,L,64]  (scale = 1/sqrt(dk))
//        1 = K store  [B,H,L,64]
//        2 = V store  [B,H,64,L]  (transposed for the P*V B-operand)
//        3 = f32 store m*1024+n with residual add (final projection)
// ---------------------------------------------------------------------------
__global__ __launch_bounds__(256) void gemm_xwT(const _Float16* __restrict__ X,
                                                const _Float16* __restrict__ W,
                                                const float* __restrict__ R,
                                                _Float16* __restrict__ outh,
                                                float* __restrict__ outf,
                                                int mode, float scale) {
  int lane = threadIdx.x & 31;
  int wave = threadIdx.x >> 5;
  int wm = wave & 3, wn = wave >> 2;
  int m0 = blockIdx.x * 128 + wm * 32;
  int n0 = blockIdx.y * 128 + wn * 64;

  v8f c[2][4] = {};
  for (int k0 = 0; k0 < D_MODEL; k0 += 32) {
    v16h a0 = load_frag(X, D_MODEL, m0,      k0, lane);
    v16h a1 = load_frag(X, D_MODEL, m0 + 16, k0, lane);
    v16h b0 = load_frag(W, D_MODEL, n0,      k0, lane);
    v16h b1 = load_frag(W, D_MODEL, n0 + 16, k0, lane);
    v16h b2 = load_frag(W, D_MODEL, n0 + 32, k0, lane);
    v16h b3 = load_frag(W, D_MODEL, n0 + 48, k0, lane);
    c[0][0] = wmma16x16x32(a0, b0, c[0][0]);
    c[0][1] = wmma16x16x32(a0, b1, c[0][1]);
    c[0][2] = wmma16x16x32(a0, b2, c[0][2]);
    c[0][3] = wmma16x16x32(a0, b3, c[0][3]);
    c[1][0] = wmma16x16x32(a1, b0, c[1][0]);
    c[1][1] = wmma16x16x32(a1, b1, c[1][1]);
    c[1][2] = wmma16x16x32(a1, b2, c[1][2]);
    c[1][3] = wmma16x16x32(a1, b3, c[1][3]);
  }

  // C layout: element r of the v8f = row (r + 8*(lane>>4)), column (lane&15).
  int col = lane & 15, hh = lane >> 4;
#pragma unroll
  for (int i = 0; i < 2; ++i)
#pragma unroll
    for (int j = 0; j < 4; ++j)
#pragma unroll
      for (int r = 0; r < 8; ++r) {
        int m = m0 + 16 * i + r + 8 * hh;
        int nn = n0 + 16 * j + col;
        float val = c[i][j][r] * scale;
        if (mode == 3) {
          size_t idx = (size_t)m * D_MODEL + nn;
          outf[idx] = val + R[idx];
        } else {
          int b = m >> 11, l = m & (LSEQ - 1);
          int hd = nn >> 6, d = nn & (DHEAD - 1);
          size_t idx;
          if (mode == 2)
            idx = ((size_t)(b * NHEAD + hd) * DHEAD + d) * LSEQ + l;
          else
            idx = ((size_t)(b * NHEAD + hd) * LSEQ + l) * DHEAD + d;
          outh[idx] = (_Float16)val;
        }
      }
}

// ---------------------------------------------------------------------------
// Flash attention. Block = 4 waves sharing one (b,h); each wave owns a 16-row
// q tile. Per key-loop iteration the block cooperatively DMAs the 32x64 K-tile
// and 64x32 V^T-tile into LDS (async, double-buffered), then each wave runs
// 8 WMMA + online softmax. Mask is all-ones in this problem.
//   Qh [B*H, L, 64] (pre-scaled by 1/8), Kh [B*H, L, 64], Vt [B*H, 64, L]
//   Oh [B, L, H*64] f16
// ---------------------------------------------------------------------------
__global__ __launch_bounds__(128) void attn_kernel(const _Float16* __restrict__ Qh,
                                                   const _Float16* __restrict__ Kh,
                                                   const _Float16* __restrict__ Vt,
                                                   _Float16* __restrict__ Oh) {
  __shared__ _Float16 Kt[2][32 * 64];   // 2 x 4 KiB K tiles (rows x d)
  __shared__ _Float16 Vl[2][64 * 32];   // 2 x 4 KiB V^T tiles (d x keys)
  __shared__ _Float16 Pb[4][16 * 32];   // per-wave P transpose buffers

  int tid = threadIdx.x;
  int lane = tid & 31;
  int wave = tid >> 5;
  int bh = blockIdx.x >> 5;                    // 32 blocks per (b,h)
  int qt = (blockIdx.x & 31) * 4 + wave;       // q tile of this wave
  int m0 = qt * 16;
  const _Float16* Qp = Qh + (size_t)bh * LSEQ * DHEAD;
  const _Float16* Kp = Kh + (size_t)bh * LSEQ * DHEAD;
  const _Float16* Vp = Vt + (size_t)bh * DHEAD * LSEQ;

  // Stage one 32-key tile: K 32x64 (256 16B chunks) + V^T 64x32 (256 chunks).
  auto stage = [&](int kb, _Float16* kt, _Float16* vl) {
#pragma unroll
    for (int s = 0; s < 2; ++s) {
      int ck = tid + 128 * s;
      int kr = ck >> 3, ko = (ck & 7) * 8;                 // K: 8 chunks/row
      async_b128(kt + kr * 64 + ko, Kp + (size_t)(kb + kr) * DHEAD + ko);
      int vr = ck >> 2, vo = (ck & 3) * 8;                 // V: 4 chunks/row
      async_b128(vl + vr * 32 + vo, Vp + (size_t)vr * LSEQ + kb + vo);
    }
  };

  v16h qa0 = load_frag(Qp, DHEAD, m0, 0, lane);
  v16h qa1 = load_frag(Qp, DHEAD, m0, 32, lane);

  v8f o0 = {}, o1 = {}, o2 = {}, o3 = {};
  float mrow[8], lrow[8];
#pragma unroll
  for (int r = 0; r < 8; ++r) { mrow[r] = -3.0e38f; lrow[r] = 0.f; }

  int col = lane & 15, hh = lane >> 4;
  _Float16* myP = &Pb[wave][0];

  stage(0, Kt[0], Vl[0]);
  int cur = 0;

  for (int kb = 0; kb < LSEQ; kb += 32) {
    asm volatile("s_wait_asynccnt 0" ::: "memory");  // this wave's DMAs landed
    __syncthreads();                                 // every wave's DMAs landed
    if (kb + 32 < LSEQ) stage(kb + 32, Kt[cur ^ 1], Vl[cur ^ 1]);

    const _Float16* kt = Kt[cur];
    const _Float16* vl = Vl[cur];

    // S[16 x 32] = Q * K^T from the LDS tile
    v8f s0 = {}, s1 = {};
    {
      v16h k00 = load_frag(kt, DHEAD, 0,  0,  lane);
      v16h k01 = load_frag(kt, DHEAD, 0,  32, lane);
      s0 = wmma16x16x32(qa0, k00, s0);
      s0 = wmma16x16x32(qa1, k01, s0);
      v16h k10 = load_frag(kt, DHEAD, 16, 0,  lane);
      v16h k11 = load_frag(kt, DHEAD, 16, 32, lane);
      s1 = wmma16x16x32(qa0, k10, s1);
      s1 = wmma16x16x32(qa1, k11, s1);
    }
    // Online softmax over the 32 new keys, per row r (row = r + 8*hh).
#pragma unroll
    for (int r = 0; r < 8; ++r) {
      float vmax = fmaxf(s0[r], s1[r]);
#pragma unroll
      for (int d = 1; d < 16; d <<= 1) vmax = fmaxf(vmax, __shfl_xor(vmax, d, 32));
      float mnew = fmaxf(mrow[r], vmax);
      float corr = __expf(mrow[r] - mnew);
      float p0 = __expf(s0[r] - mnew);
      float p1 = __expf(s1[r] - mnew);
      float rs = p0 + p1;
#pragma unroll
      for (int d = 1; d < 16; d <<= 1) rs += __shfl_xor(rs, d, 32);
      lrow[r] = lrow[r] * corr + rs;
      mrow[r] = mnew;
      o0[r] *= corr; o1[r] *= corr; o2[r] *= corr; o3[r] *= corr;
      // Transpose P (C layout -> row-major 16x32) through LDS.
      myP[(r + 8 * hh) * 32 + col]      = (_Float16)p0;
      myP[(r + 8 * hh) * 32 + 16 + col] = (_Float16)p1;
    }
    asm volatile("s_wait_dscnt 0" ::: "memory");  // intra-wave LDS RAW fence
    v16h pa  = load_frag(myP, 32, 0,  0, lane);   // P as A-operand (16x32)
    v16h vb0 = load_frag(vl, 32, 0,  0, lane);    // V^T rows = d, K = keys
    v16h vb1 = load_frag(vl, 32, 16, 0, lane);
    v16h vb2 = load_frag(vl, 32, 32, 0, lane);
    v16h vb3 = load_frag(vl, 32, 48, 0, lane);
    o0 = wmma16x16x32(pa, vb0, o0);
    o1 = wmma16x16x32(pa, vb1, o1);
    o2 = wmma16x16x32(pa, vb2, o2);
    o3 = wmma16x16x32(pa, vb3, o3);
    cur ^= 1;
  }

  int b = bh >> 4, h = bh & (NHEAD - 1);
#pragma unroll
  for (int r = 0; r < 8; ++r) {
    float inv = 1.0f / lrow[r];
    size_t m = (size_t)(b * LSEQ) + m0 + r + 8 * hh;
    size_t base = m * D_MODEL + h * DHEAD + col;
    Oh[base + 0]  = (_Float16)(o0[r] * inv);
    Oh[base + 16] = (_Float16)(o1[r] * inv);
    Oh[base + 32] = (_Float16)(o2[r] * inv);
    Oh[base + 48] = (_Float16)(o3[r] * inv);
  }
}

// ---------------------------------------------------------------------------
// Host-side orchestration
// ---------------------------------------------------------------------------
extern "C" void kernel_launch(void* const* d_in, const int* in_sizes, int n_in,
                              void* d_out, int out_size, void* d_ws, size_t ws_size,
                              hipStream_t stream) {
  (void)in_sizes; (void)n_in; (void)out_size; (void)ws_size;
  const float* q   = (const float*)d_in[0];
  const float* k   = (const float*)d_in[1];
  const float* v   = (const float*)d_in[2];
  /* d_in[3] = mask (all ones; unused) */
  const float* wq  = (const float*)d_in[4];
  const float* wk  = (const float*)d_in[5];
  const float* wv  = (const float*)d_in[6];
  const float* wo  = (const float*)d_in[7];
  const float* lng = (const float*)d_in[8];
  const float* lnb = (const float*)d_in[9];
  float* out = (float*)d_out;

  char* ws = (char*)d_ws;
  const size_t MiB = 1u << 20;
  _Float16* qn_h = (_Float16*)(ws + 0 * MiB);   // 8 MiB  LN(q) f16 [4096,1024]
  _Float16* kh   = (_Float16*)(ws + 8 * MiB);   // 8 MiB  k f16
  _Float16* vh   = (_Float16*)(ws + 16 * MiB);  // 8 MiB  v f16
  _Float16* wqh  = (_Float16*)(ws + 24 * MiB);  // 2 MiB
  _Float16* wkh  = (_Float16*)(ws + 26 * MiB);  // 2 MiB
  _Float16* wvh  = (_Float16*)(ws + 28 * MiB);  // 2 MiB
  _Float16* woh  = (_Float16*)(ws + 30 * MiB);  // 2 MiB
  _Float16* Qh   = (_Float16*)(ws + 32 * MiB);  // 8 MiB  [B,H,L,64], pre-scaled
  _Float16* Kh   = (_Float16*)(ws + 40 * MiB);  // 8 MiB  [B,H,L,64]
  _Float16* Vt   = (_Float16*)(ws + 48 * MiB);  // 8 MiB  [B,H,64,L]
  _Float16* Ah   = (_Float16*)(ws + 56 * MiB);  // 8 MiB  attn out [B,L,1024]

  // 1) LayerNorm + downconversions
  ln_kernel<<<NROWS, 256, 0, stream>>>(q, lng, lnb, qn_h);
  int nact = NROWS * D_MODEL;
  f32_to_f16<<<(nact + 255) / 256, 256, 0, stream>>>(k, kh, nact);
  f32_to_f16<<<(nact + 255) / 256, 256, 0, stream>>>(v, vh, nact);
  int nw = D_MODEL * D_MODEL;
  f32_to_f16<<<(nw + 255) / 256, 256, 0, stream>>>(wq, wqh, nw);
  f32_to_f16<<<(nw + 255) / 256, 256, 0, stream>>>(wk, wkh, nw);
  f32_to_f16<<<(nw + 255) / 256, 256, 0, stream>>>(wv, wvh, nw);
  f32_to_f16<<<(nw + 255) / 256, 256, 0, stream>>>(wo, woh, nw);

  // 2) QKV projections (WMMA GEMMs); Q pre-scaled by 1/sqrt(64)=0.125
  dim3 gg(NROWS / 128, D_MODEL / 128);
  gemm_xwT<<<gg, 256, 0, stream>>>(qn_h, wqh, nullptr, Qh, nullptr, 0, 0.125f);
  gemm_xwT<<<gg, 256, 0, stream>>>(kh,   wkh, nullptr, Kh, nullptr, 1, 1.0f);
  gemm_xwT<<<gg, 256, 0, stream>>>(vh,   wvh, nullptr, Vt, nullptr, 2, 1.0f);

  // 3) Flash attention (4096 q-tiles, 4 waves / block, shared K/V staging)
  attn_kernel<<<(BATCH * NHEAD * (LSEQ / 16)) / 4, 128, 0, stream>>>(Qh, Kh, Vt, Ah);

  // 4) Output projection + residual (f32 out)
  gemm_xwT<<<gg, 256, 0, stream>>>(Ah, woh, q, nullptr, out, 3, 1.0f);
}
